// CSDeformAttn_44916767982055
// MI455X (gfx1250) — matrix operand order repaired and checked by reading, not compiled, not run on hardware
//
#include <hip/hip_runtime.h>
#include <hip/hip_bf16.h>
#include <math.h>

// ---------------- CDNA5 WMMA plumbing (gfx1250, wave32) ----------------
typedef __attribute__((ext_vector_type(16))) __bf16 v16bf;
typedef __attribute__((ext_vector_type(8)))  __bf16 v8bf;
typedef __attribute__((ext_vector_type(8)))  float  v8f;

#define DEV __device__ __forceinline__

// Problem geometry (fixed by the reference)
#define NB  2
#define NQ  1024
#define DM  256
#define NH  8
#define DH  64
#define NL  4
#define NP  4
#define NM  (NL*NP)   // 16

DEV v16bf make16(v8bf lo, v8bf hi) {
  v16bf f;
#pragma unroll
  for (int i = 0; i < 8; ++i) { f[i] = lo[i]; f[i + 8] = hi[i]; }
  return f;
}

// A-matrix 16x32 bf16 fragment. tile = pointer at (row0, k0); row-major, ld elems.
// lane<16: K runs {0..7, 16..23}; lane>=16: {8..15, 24..31} (ISA 7.12.2).
DEV v16bf load_fragA(const __bf16* tile, int ld, int lane) {
  const __bf16* p = tile + (size_t)(lane & 15) * ld + ((lane >> 4) << 3);
  v8bf lo = *(const v8bf*)p;
  v8bf hi = *(const v8bf*)(p + 16);
  return make16(lo, hi);
}

// B-matrix 32x16 bf16 fragment, weights stored transposed: row n holds K contiguously.
// lane<16: K runs {0..7, 16..23}; lane>=16: {8..15, 24..31} (matches A).
DEV v16bf load_fragB(const __bf16* tile, int ld, int lane) {
  const __bf16* p = tile + (size_t)(lane & 15) * ld + ((lane >> 4) << 3);
  v8bf lo = *(const v8bf*)p;
  v8bf hi = *(const v8bf*)(p + 16);
  return make16(lo, hi);
}

DEV v8f wmma_bf16(v16bf a, v16bf b, v8f c) {
  return __builtin_amdgcn_wmma_f32_16x16x32_bf16(false, a, false, b, (short)0, c,
                                                 false, false);
}

DEV v8f zero8() { v8f a; 
#pragma unroll
  for (int i = 0; i < 8; ++i) a[i] = 0.f; return a; }

// ---------------- prep kernels ----------------
__global__ __launch_bounds__(256) void k_cast(const float* __restrict__ s,
                                              __bf16* __restrict__ d, int n) {
  int i = blockIdx.x * 256 + threadIdx.x;
  if (i < n) d[i] = (__bf16)s[i];
}

// W_k (8,64,256) -> WkT (8,256,64) bf16 : rows over c hold contiguous d (=K of qk GEMM)
__global__ __launch_bounds__(256) void k_wk_t(const float* __restrict__ wk,
                                              __bf16* __restrict__ wkt) {
  int i = blockIdx.x * 256 + threadIdx.x;
  if (i < NH * DM * DH) {
    int d = i & (DH - 1);
    int c = (i >> 6) & (DM - 1);
    int h = i >> 14;
    wkt[i] = (__bf16)wk[((size_t)h * DH + d) * DM + c];
  }
}

// ---------------- GEMM 1: off (f32) + q (bf16) = iq @ [W_off^T | W_q^T] ------
__global__ __launch_bounds__(256) void k_gemm_off_q(
    const __bf16* __restrict__ iq, const __bf16* __restrict__ woff,
    const __bf16* __restrict__ wq, const float* __restrict__ boff,
    const float* __restrict__ bq, float* __restrict__ off_out,
    __bf16* __restrict__ q_out) {
  int lane = threadIdx.x & 31, wv = threadIdx.x >> 5;
  int m0 = blockIdx.x * 16;
  int ntile = blockIdx.y * 8 + wv;          // 0..47 ; 0..15 -> off, 16..47 -> q
  int n0 = ntile * 16;
  bool is_off = (n0 < DM);
  const __bf16* B = is_off ? (woff + (size_t)n0 * DM)
                           : (wq + (size_t)(n0 - DM) * DM);
  v8f acc = zero8();
#pragma unroll
  for (int k = 0; k < DM; k += 32) {
    v16bf a = load_fragA(iq + (size_t)m0 * DM + k, DM, lane);
    v16bf b = load_fragB(B + k, DM, lane);
    acc = wmma_bf16(a, b, acc);
  }
  int col = lane & 15, rb = (lane >> 4) << 3;
  if (is_off) {
    float bias = boff[n0 + col];
#pragma unroll
    for (int r = 0; r < 8; ++r)
      off_out[(size_t)(m0 + rb + r) * DM + n0 + col] = acc[r] + bias;
  } else {
    int nq = n0 - DM;
    float bias = bq[nq + col];
#pragma unroll
    for (int r = 0; r < 8; ++r)
      q_out[(size_t)(m0 + rb + r) * (NH * DH) + nq + col] = (__bf16)(acc[r] + bias);
  }
}

// ---------------- GEMM 2: qk[n,h,c] = scale * q[n,h,:] @ W_k[h] ----------------
__global__ __launch_bounds__(256) void k_gemm_qk(
    const __bf16* __restrict__ qbf, const __bf16* __restrict__ wkt,
    float* __restrict__ qk, float scale) {
  int lane = threadIdx.x & 31, wv = threadIdx.x >> 5;
  int m0 = blockIdx.x * 16;
  int h = blockIdx.z;
  int n0 = (blockIdx.y * 8 + wv) * 16;       // c tile
  const __bf16* A = qbf + (size_t)m0 * (NH * DH) + h * DH;
  const __bf16* B = wkt + ((size_t)h * DM + n0) * DH;
  v8f acc = zero8();
#pragma unroll
  for (int k = 0; k < DH; k += 32) {
    v16bf a = load_fragA(A + k, NH * DH, lane);
    v16bf b = load_fragB(B + k, DH, lane);
    acc = wmma_bf16(a, b, acc);
  }
  int col = lane & 15, rb = (lane >> 4) << 3;
#pragma unroll
  for (int r = 0; r < 8; ++r)
    qk[((size_t)(m0 + rb + r) * NH + h) * DM + n0 + col] = acc[r] * scale;
}

// ---------------- fused sampling + attention + W_v projection ----------------
DEV void sample_point(const __bf16* __restrict__ flat_bf, int b, int len_in,
                      int Hl, int Wl, int Sl, float ix, float iy, int lane,
                      float f[8]) {
  float x0f = floorf(ix), y0f = floorf(iy);
  int x0 = (int)x0f, y0 = (int)y0f;
  float wx1 = ix - x0f, wy1 = iy - y0f;
  float wx0 = 1.f - wx1, wy0 = 1.f - wy1;
  float w[4] = {wx0 * wy0, wx1 * wy0, wx0 * wy1, wx1 * wy1};
  int xs[4] = {x0, x0 + 1, x0, x0 + 1};
  int ys[4] = {y0, y0, y0 + 1, y0 + 1};
  const __bf16* base = flat_bf + (size_t)b * len_in * DM + lane * 8;
#pragma unroll
  for (int j = 0; j < 8; ++j) f[j] = 0.f;
#pragma unroll
  for (int q = 0; q < 4; ++q) {
    int xc = xs[q], yc = ys[q];
    if (xc >= 0 && xc < Wl && yc >= 0 && yc < Hl) {
      const v8bf v = *(const v8bf*)(base + (size_t)(Sl + yc * Wl + xc) * DM);
      float wq = w[q];
#pragma unroll
      for (int j = 0; j < 8; ++j) f[j] += wq * (float)v[j];
    }
  }
}

__global__ __launch_bounds__(256) void k_sample_attend(
    const __bf16* __restrict__ flat_bf, const float* __restrict__ ref,
    const float* __restrict__ off, const float* __restrict__ qk,
    const int* __restrict__ shapes, const int* __restrict__ starts,
    const __bf16* __restrict__ wv_w, const float* __restrict__ bv,
    __bf16* __restrict__ head_out, int len_in) {
  __shared__ float s_ix[16][NM];
  __shared__ float s_iy[16][NM];
  __shared__ float s_dt[16][NM];
  __shared__ int s_H[NL], s_W[NL], s_S[NL];
  __shared__ __bf16 s_agg[16][DM];

  int t = threadIdx.x, lane = t & 31, wid = t >> 5;
  int b = blockIdx.z, h = blockIdx.y, n0 = blockIdx.x * 16;

  if (t < NL) { s_H[t] = shapes[2 * t]; s_W[t] = shapes[2 * t + 1]; s_S[t] = starts[t]; }
  __syncthreads();

  // phase 0: sampling coordinates (one thread per (query, m))
  {
    int n = t >> 4, m = t & 15, l = m >> 2, p = m & 3;
    int gq = b * NQ + n0 + n;
    float rx = ref[((size_t)gq * NL + l) * 2 + 0];
    float ry = ref[((size_t)gq * NL + l) * 2 + 1];
    int oidx = ((h * NL + l) * NP + p) * 2;
    float ox = off[(size_t)gq * DM + oidx + 0];
    float oy = off[(size_t)gq * DM + oidx + 1];
    float Wl = (float)s_W[l], Hl = (float)s_H[l];
    float gx = rx + ox / Wl, gy = ry + oy / Hl;
    s_ix[n][m] = ((gx + 1.f) * Wl - 1.f) * 0.5f;
    s_iy[n][m] = ((gy + 1.f) * Hl - 1.f) * 0.5f;
  }
  __syncthreads();

  // phase 1: dots[n][m] = feat(n,m) . qk(n)   (q.b_k constant cancels in softmax)
  for (int idx = wid; idx < 16 * NM; idx += 8) {
    int n = idx >> 4, m = idx & 15, l = m >> 2;
    int gq = b * NQ + n0 + n;
    const float* qkp = qk + ((size_t)gq * NH + h) * DM + lane * 8;
    float4 qa = ((const float4*)qkp)[0];
    float4 qb = ((const float4*)qkp)[1];
    float qv[8] = {qa.x, qa.y, qa.z, qa.w, qb.x, qb.y, qb.z, qb.w};
    float f[8];
    sample_point(flat_bf, b, len_in, s_H[l], s_W[l], s_S[l],
                 s_ix[n][m], s_iy[n][m], lane, f);
    float d = 0.f;
#pragma unroll
    for (int j = 0; j < 8; ++j) d += f[j] * qv[j];
#pragma unroll
    for (int o = 16; o >= 1; o >>= 1) d += __shfl_xor(d, o, 32);
    if (lane == 0) s_dt[n][m] = d;
  }
  __syncthreads();

  // phase 2: softmax over the 16 (level,point) samples
  if (t < 16) {
    float mx = -3.0e38f;
#pragma unroll
    for (int m = 0; m < NM; ++m) mx = fmaxf(mx, s_dt[t][m]);
    float e[NM], s = 0.f;
#pragma unroll
    for (int m = 0; m < NM; ++m) { e[m] = __expf(s_dt[t][m] - mx); s += e[m]; }
    float inv = 1.f / s;
#pragma unroll
    for (int m = 0; m < NM; ++m) s_dt[t][m] = e[m] * inv;
  }
  __syncthreads();

  // phase 3: attention-weighted feature aggregate (re-gather: hits L0/L2)
  for (int n = wid; n < 16; n += 8) {
    float acc[8];
#pragma unroll
    for (int j = 0; j < 8; ++j) acc[j] = 0.f;
    for (int m = 0; m < NM; ++m) {
      int l = m >> 2;
      float f[8];
      sample_point(flat_bf, b, len_in, s_H[l], s_W[l], s_S[l],
                   s_ix[n][m], s_iy[n][m], lane, f);
      float a = s_dt[n][m];
#pragma unroll
      for (int j = 0; j < 8; ++j) acc[j] += a * f[j];
    }
#pragma unroll
    for (int j = 0; j < 8; ++j) s_agg[n][lane * 8 + j] = (__bf16)acc[j];
  }
  __syncthreads();

  // phase 4: head_out = agg @ W_v[h]^T + b_v  (WMMA, 4 waves x 16x16 tiles)
  if (wid < 4) {
    int d0 = wid * 16;
    const __bf16* B = wv_w + ((size_t)h * DH + d0) * DM;
    v8f acc = zero8();
#pragma unroll
    for (int k = 0; k < DM; k += 32) {
      v16bf a = load_fragA(&s_agg[0][0] + k, DM, lane);
      v16bf bb = load_fragB(B + k, DM, lane);
      acc = wmma_bf16(a, bb, acc);
    }
    int col = lane & 15, rb = (lane >> 4) << 3;
    float bias = bv[h * DH + d0 + col];
#pragma unroll
    for (int r = 0; r < 8; ++r) {
      int gq = b * NQ + n0 + rb + r;
      head_out[(size_t)gq * (NH * DH) + h * DH + d0 + col] = (__bf16)(acc[r] + bias);
    }
  }
}

// ---------------- GEMM 3: out = head_out @ W_o^T + b_o (f32 result) ----------
__global__ __launch_bounds__(256) void k_gemm_out(
    const __bf16* __restrict__ head, const __bf16* __restrict__ wo,
    const float* __restrict__ bo, float* __restrict__ out) {
  int lane = threadIdx.x & 31, wv = threadIdx.x >> 5;
  int m0 = blockIdx.x * 16;
  int n0 = (blockIdx.y * 8 + wv) * 16;
  const __bf16* B = wo + (size_t)n0 * (NH * DH);
  v8f acc = zero8();
#pragma unroll
  for (int k = 0; k < NH * DH; k += 32) {
    v16bf a = load_fragA(head + (size_t)m0 * (NH * DH) + k, NH * DH, lane);
    v16bf b = load_fragB(B + k, NH * DH, lane);
    acc = wmma_bf16(a, b, acc);
  }
  int col = lane & 15, rb = (lane >> 4) << 3;
  float bias = bo[n0 + col];
#pragma unroll
  for (int r = 0; r < 8; ++r)
    out[(size_t)(m0 + rb + r) * DM + n0 + col] = acc[r] + bias;
}

// ---------------- host launcher ----------------
extern "C" void kernel_launch(void* const* d_in, const int* in_sizes, int n_in,
                              void* d_out, int out_size, void* d_ws, size_t ws_size,
                              hipStream_t stream) {
  const float* iq_f   = (const float*)d_in[0];
  const float* ref    = (const float*)d_in[1];
  const float* flat_f = (const float*)d_in[2];
  const int*   shapes = (const int*)d_in[3];
  const int*   starts = (const int*)d_in[4];
  const float* Woff   = (const float*)d_in[5];
  const float* boff   = (const float*)d_in[6];
  const float* Wq     = (const float*)d_in[7];
  const float* bq     = (const float*)d_in[8];
  const float* Wk     = (const float*)d_in[9];
  /* d_in[10] = b_k: provably cancels in softmax -> unused */
  const float* Wv     = (const float*)d_in[11];
  const float* bv     = (const float*)d_in[12];
  const float* Wo     = (const float*)d_in[13];
  const float* bo     = (const float*)d_in[14];

  const int len_in = in_sizes[2] / (NB * DM);   // 44965
  const int n_flat = in_sizes[2];               // NB*len_in*DM

  char* ws = (char*)d_ws;
  size_t pos = 0;
  auto alloc = [&](size_t bytes) -> void* {
    void* p = ws + pos;
    pos = (pos + bytes + 255) & ~(size_t)255;
    return p;
  };
  __bf16* flat_bf = (__bf16*)alloc((size_t)n_flat * 2);
  __bf16* iq_bf   = (__bf16*)alloc((size_t)NB * NQ * DM * 2);
  __bf16* woff_bf = (__bf16*)alloc((size_t)DM * DM * 2);
  __bf16* wq_bf   = (__bf16*)alloc((size_t)NH * DH * DM * 2);
  __bf16* wkt_bf  = (__bf16*)alloc((size_t)NH * DM * DH * 2);
  __bf16* wv_bf   = (__bf16*)alloc((size_t)NH * DH * DM * 2);
  __bf16* wo_bf   = (__bf16*)alloc((size_t)DM * NH * DH * 2);
  float*  off_f   = (float*)alloc((size_t)NB * NQ * DM * 4);
  __bf16* q_bf    = (__bf16*)alloc((size_t)NB * NQ * NH * DH * 2);
  float*  qk_f    = (float*)alloc((size_t)NB * NQ * NH * DM * 4);
  __bf16* head_bf = (__bf16*)alloc((size_t)NB * NQ * NH * DH * 2);

  auto cast = [&](const float* s, __bf16* d, int n) {
    k_cast<<<(n + 255) / 256, 256, 0, stream>>>(s, d, n);
  };
  cast(flat_f, flat_bf, n_flat);
  cast(iq_f, iq_bf, NB * NQ * DM);
  cast(Woff, woff_bf, DM * DM);
  cast(Wq, wq_bf, NH * DH * DM);
  cast(Wv, wv_bf, NH * DH * DM);
  cast(Wo, wo_bf, DM * NH * DH);
  k_wk_t<<<(NH * DM * DH + 255) / 256, 256, 0, stream>>>(Wk, wkt_bf);

  // off + q projections: (2048 x 256) x (256 x 768)
  k_gemm_off_q<<<dim3(NB * NQ / 16, 6), 256, 0, stream>>>(
      iq_bf, woff_bf, wq_bf, boff, bq, off_f, q_bf);

  // qk = scale * q_h @ W_k[h] : per head (2048 x 64) x (64 x 256)
  k_gemm_qk<<<dim3(NB * NQ / 16, 2, NH), 256, 0, stream>>>(
      q_bf, wkt_bf, qk_f, 0.125f);

  // fused bilinear sampling + softmax attention + W_v projection
  k_sample_attend<<<dim3(NQ / 16, NH, NB), 256, 0, stream>>>(
      flat_bf, ref, off_f, qk_f, shapes, starts, wv_bf, bv, head_bf, len_in);

  // final output projection: (2048 x 512) x (512 x 256) + b_o
  k_gemm_out<<<dim3(NB * NQ / 16, 2), 256, 0, stream>>>(
      head_bf, wo_bf, bo, (float*)d_out);
}
